// GraphPyramidModule_65867618451914
// MI455X (gfx1250) — matrix operand order repaired and checked by reading, not compiled, not run on hardware
//
#include <hip/hip_runtime.h>
#include <hip/hip_bf16.h>

// ---------------------------------------------------------------------------
// Types for CDNA5 WMMA (wave32, V_WMMA_F32_16X16X32_BF16)
// ---------------------------------------------------------------------------
typedef __attribute__((ext_vector_type(16))) __bf16         bf16x16;
typedef __attribute__((ext_vector_type(8)))  float          f32x8;
typedef __attribute__((ext_vector_type(8)))  unsigned short u16x8;

static __device__ __forceinline__ unsigned short f32_to_bf16_bits(float f) {
    unsigned int u = __float_as_uint(f);
    u += 0x7fffu + ((u >> 16) & 1u);      // round-to-nearest-even
    return (unsigned short)(u >> 16);
}
static __device__ __forceinline__ unsigned int pack_bf16x2(float lo, float hi) {
    return (unsigned int)f32_to_bf16_bits(lo) |
           ((unsigned int)f32_to_bf16_bits(hi) << 16);
}

static __device__ __forceinline__ void atomicMaxF(float* addr, float val) {
    unsigned int* ua = (unsigned int*)addr;
    unsigned int  old = *ua;
    while (__uint_as_float(old) < val) {
        unsigned int assumed = old;
        old = atomicCAS(ua, assumed, __float_as_uint(val));
        if (old == assumed) break;
    }
}

#define NEG_INF (-__builtin_inff())

// ---------------------------------------------------------------------------
// 1) semantic 1x1 conv (C=256 -> 20) + argmax; writes x_aux and raw mask
// ---------------------------------------------------------------------------
__global__ __launch_bounds__(256)
void sem_argmax_kernel(const float* __restrict__ feat, const float* __restrict__ semW,
                       const float* __restrict__ semB, float* __restrict__ xaux,
                       int* __restrict__ mask, int hw, int total) {
    int idx = blockIdx.x * blockDim.x + threadIdx.x;
    if (idx >= total) return;
    int b = idx / hw, p = idx - b * hw;
    float acc[20];
#pragma unroll
    for (int k = 0; k < 20; ++k) acc[k] = semB[k];
    for (int c = 0; c < 256; ++c) {
        float f = feat[((size_t)b * 256 + c) * hw + p];
#pragma unroll
        for (int k = 0; k < 20; ++k) acc[k] += semW[k * 256 + c] * f;
    }
    int best = 0; float bv = acc[0];
#pragma unroll
    for (int k = 1; k < 20; ++k) if (acc[k] > bv) { bv = acc[k]; best = k; }
    mask[idx] = best;
#pragma unroll
    for (int k = 0; k < 20; ++k) xaux[((size_t)b * 20 + k) * hw + p] = acc[k];
}

// ---------------------------------------------------------------------------
// 2) aggregation: per-(batch,label) channel sum / masked-max / count
// ---------------------------------------------------------------------------
__global__ __launch_bounds__(256)
void agg_init_kernel(float* gsum, float* gmax, int* gcnt, int totKC, int totK) {
    int i = blockIdx.x * blockDim.x + threadIdx.x;
    if (i < totKC) { gsum[i] = 0.f; gmax[i] = NEG_INF; }
    if (i < totK) gcnt[i] = 0;
}

#define AGG_PIX 512
__global__ __launch_bounds__(256)
void agg_kernel(const float* __restrict__ feat, const int* __restrict__ mask, int labDiv,
                int K, float* gsum, float* gmax, int* gcnt, int hw) {
    const int b = blockIdx.z;
    const int pBase = blockIdx.x * AGG_PIX;
    const int tid = threadIdx.x;              // tid == channel (0..255)

    __shared__ float lsum[20][256];
    __shared__ float lmax[20][256];
    __shared__ int   lcnt[20];
    __shared__ float tile[256][17];           // padded: bank-conflict-free [c][p]
    __shared__ int   slab[16];

    for (int k = 0; k < K; ++k) { lsum[k][tid] = 0.f; lmax[k][tid] = NEG_INF; }
    if (tid < 20) lcnt[tid] = 0;
    __syncthreads();

    for (int ch = 0; ch < AGG_PIX; ch += 16) {
        if (tid < 16) {
            int lab = mask[b * hw + pBase + ch + tid] / labDiv;
            slab[tid] = lab;
            atomicAdd(&lcnt[lab], 1);
        }
        // coalesced stage: 16 pixels x 256 channels
#pragma unroll
        for (int i = 0; i < 16; ++i) {
            int e = tid + 256 * i;
            int p = e & 15, c = e >> 4;
            tile[c][p] = feat[((size_t)b * 256 + c) * hw + pBase + ch + p];
        }
        __syncthreads();
#pragma unroll
        for (int p = 0; p < 16; ++p) {        // thread owns channel tid: race-free
            int lab = slab[p];
            float v = tile[tid][p];
            lsum[lab][tid] += v;
            lmax[lab][tid] = fmaxf(lmax[lab][tid], v);
        }
        __syncthreads();
    }
    for (int k = 0; k < K; ++k) {
        atomicAdd(&gsum[((size_t)b * K + k) * 256 + tid], lsum[k][tid]);
        atomicMaxF(&gmax[((size_t)b * K + k) * 256 + tid], lmax[k][tid]);
    }
    if (tid < K) atomicAdd(&gcnt[b * K + tid], lcnt[tid]);
}

// node features X[b,k,0:256]=mean, X[b,k,256:512]=masked max (0 competes, per ref)
__global__ __launch_bounds__(256)
void finalize_nodes_kernel(const float* gsum, const float* gmax, const int* gcnt,
                           float* X, int K, int hw, int total) {
    int i = blockIdx.x * blockDim.x + threadIdx.x;
    if (i >= total) return;
    int c2 = i & 511; int nk = i >> 9; int k = nk % K; int b = nk / K;
    int cnt = gcnt[b * K + k];
    float v;
    if (c2 < 256) {
        v = gsum[((size_t)b * K + k) * 256 + c2] / (float)(cnt > 0 ? cnt : 1);
    } else {
        float m = gmax[((size_t)b * K + k) * 256 + (c2 - 256)];
        v = (cnt == hw) ? m : fmaxf(m, 0.f);  // zeros from non-members compete
    }
    X[(size_t)(b * K + k) * 512 + c2] = v;
}

// ---------------------------------------------------------------------------
// 3) tiny graph-node kernels
// ---------------------------------------------------------------------------
// OUT[b,o,k] = W[o,:] . in + bias. layout 0: in = (b,K,Cin); layout 1: in = (b,Cin,K)
__global__ __launch_bounds__(256)
void node_gemm_kernel(const float* __restrict__ in, const float* __restrict__ W,
                      const float* __restrict__ bias, float* __restrict__ out,
                      int Cin, int Cout, int K, int layout, int total) {
    int i = blockIdx.x * blockDim.x + threadIdx.x;
    if (i >= total) return;
    int k = i % K; int o = (i / K) % Cout; int b = i / (K * Cout);
    float acc = bias[o];
    const float* wr = W + (size_t)o * Cin;
    if (layout == 0) {
        const float* xr = in + (size_t)(b * K + k) * Cin;
        for (int c = 0; c < Cin; ++c) acc += wr[c] * xr[c];
    } else {
        const float* xr = in + (size_t)b * Cin * K + k;
        for (int c = 0; c < Cin; ++c) acc += wr[c] * xr[(size_t)c * K];
    }
    out[((size_t)b * Cout + o) * K + k] = acc;
}

// training-mode BN over (batch,node) dims; 512 channels, h/hn layout (b,512,K)
__global__ __launch_bounds__(256)
void bn_kernel(const float* __restrict__ h, const float* __restrict__ g,
               const float* __restrict__ beta, float* __restrict__ hn, int K, int n) {
    int ch = blockIdx.x * blockDim.x + threadIdx.x;
    if (ch >= 512) return;
    int cnt = n * K;
    float m = 0.f;
    for (int b = 0; b < n; ++b)
        for (int k = 0; k < K; ++k) m += h[((size_t)b * 512 + ch) * K + k];
    m /= (float)cnt;
    float var = 0.f;
    for (int b = 0; b < n; ++b)
        for (int k = 0; k < K; ++k) {
            float d = h[((size_t)b * 512 + ch) * K + k] - m; var += d * d;
        }
    var /= (float)cnt;
    float sc = g[ch] * rsqrtf(var + 1e-5f);
    for (int b = 0; b < n; ++b)
        for (int k = 0; k < K; ++k) {
            size_t idx = ((size_t)b * 512 + ch) * K + k;
            hn[idx] = (h[idx] - m) * sc + beta[ch];
        }
}

// graph self-attention: softmax(q^T k) applied to v; h = relu(gamma*out + hn)
__global__ __launch_bounds__(256)
void attn_kernel(const float* __restrict__ q, const float* __restrict__ kk,
                 const float* __restrict__ v, const float* __restrict__ hn,
                 const float* __restrict__ gamma, float* __restrict__ hOut, int K) {
    int b = blockIdx.x; int tid = threadIdx.x;
    __shared__ float qs[128][20];
    __shared__ float ks[128][20];
    __shared__ float att[20][20];
    for (int i = tid; i < 128 * K; i += 256) {
        int k = i % K, c = i / K;
        qs[c][k] = q[((size_t)b * 128 + c) * K + k];
        ks[c][k] = kk[((size_t)b * 128 + c) * K + k];
    }
    __syncthreads();
    for (int i = tid; i < K * K; i += 256) {
        int jj = i % K, ii = i / K;
        float s = 0.f;
        for (int c = 0; c < 128; ++c) s += qs[c][ii] * ks[c][jj];
        att[ii][jj] = s;
    }
    __syncthreads();
    if (tid < K) {
        float mx = att[tid][0];
        for (int j = 1; j < K; ++j) mx = fmaxf(mx, att[tid][j]);
        float sum = 0.f;
        for (int j = 0; j < K; ++j) { float e = __expf(att[tid][j] - mx); att[tid][j] = e; sum += e; }
        float inv = 1.f / sum;
        for (int j = 0; j < K; ++j) att[tid][j] *= inv;
    }
    __syncthreads();
    float gam = gamma[0];
    for (int i = tid; i < 512 * K; i += 256) {
        int ii = i % K, c = i / K;
        const float* vr = v + ((size_t)b * 512 + c) * K;
        float s = 0.f;
        for (int j = 0; j < K; ++j) s += vr[j] * att[ii][j];
        size_t idx = ((size_t)b * 512 + c) * K + ii;
        hOut[idx] = fmaxf(gam * s + hn[idx], 0.f);
    }
}

// ---------------------------------------------------------------------------
// 4) WMMA 1x1 conv: OUT[b,o,p] = W[o,:] . A[:,p] (+bias)(+resid)(relu?)
//    Block tile: 64 pixels x 256 outch; 8 waves each own 32px x 64oc
//    = 8 v_wmma_f32_16x16x32_bf16 per wave per K-step. Activations are the
//    HBM stream (weights are L2-resident), so N=256 per block minimizes
//    A re-reads (1x for Cout=256, 2x for Cout=512).
//    A column = segmented pixel features (up to 4 tensors of Cseg channels)
//    or gathered node features g[b, c, label(p)] (gather mode, mask != null).
// ---------------------------------------------------------------------------
__global__ __launch_bounds__(256)
void conv1x1_wmma_kernel(const float* __restrict__ in0, const float* __restrict__ in1,
                         const float* __restrict__ in2, const float* __restrict__ in3,
                         int Cin, int Cseg,
                         const float* __restrict__ W, const float* __restrict__ bias, int Cout,
                         const float* __restrict__ resid, float* __restrict__ out, int do_relu,
                         const int* __restrict__ mask, int labDiv, int Kdim, int hw) {
    const int b = blockIdx.z;
    const int pTile = blockIdx.x * 64;
    const int oTile = blockIdx.y * 256;
    const int tid = threadIdx.x;
    const int lane = tid & 31, wave = tid >> 5;
    const int mW = (wave & 1) * 32;           // wave pixel offset   (2 waves in M)
    const int nW = (wave >> 1) * 64;          // wave outch offset   (4 waves in N)

    __shared__ unsigned short As[64][40];     // [pixel][k] bf16, padded row (80B)
    __shared__ unsigned short Bs[256][40];    // [outch][k] bf16, padded row
    __shared__ int labS[64];

    if (mask && tid < 64) labS[tid] = mask[b * hw + pTile + tid] / labDiv;
    __syncthreads();

    f32x8 acc[2][4];
#pragma unroll
    for (int mt = 0; mt < 2; ++mt)
#pragma unroll
        for (int nt = 0; nt < 4; ++nt)
#pragma unroll
            for (int j = 0; j < 8; ++j) acc[mt][nt][j] = 0.f;

    for (int kb = 0; kb < Cin; kb += 32) {
        // ---- stage A: 64 px x 32 ch as packed bf16x2 (1024 pairs, 4/thread) ----
        if (mask) {
            const float* g = in0;             // (b, Cin, Kdim)
#pragma unroll
            for (int i = 0; i < 4; ++i) {
                int e = tid + 256 * i;
                int p = e & 63, c = (e >> 6) * 2;
                float v0 = g[((size_t)b * Cin + kb + c) * Kdim + labS[p]];
                float v1 = g[((size_t)b * Cin + kb + c + 1) * Kdim + labS[p]];
                *(unsigned int*)&As[p][c] = pack_bf16x2(v0, v1);
            }
        } else {
            int segIdx = kb / Cseg;           // Cseg multiple of 32: no straddle
            int cOff = kb - segIdx * Cseg;
            const float* seg = (segIdx == 0) ? in0 : (segIdx == 1) ? in1
                             : (segIdx == 2) ? in2 : in3;
            const float* basep = seg + ((size_t)b * Cseg + cOff) * hw + pTile;
            __builtin_prefetch(basep + 32 * (size_t)hw, 0, 1);
#pragma unroll
            for (int i = 0; i < 4; ++i) {
                int e = tid + 256 * i;
                int p = e & 63, c = (e >> 6) * 2;
                float v0 = basep[(size_t)c * hw + p];
                float v1 = basep[(size_t)(c + 1) * hw + p];
                *(unsigned int*)&As[p][c] = pack_bf16x2(v0, v1);
            }
        }
        // ---- stage B: 256 oc x 32 ch as packed bf16x2 (4096 pairs, 16/thread) ----
#pragma unroll
        for (int i = 0; i < 16; ++i) {
            int e = tid + 256 * i;
            int c = (e & 15) * 2, o = e >> 4;
            const float* wr = W + (size_t)(oTile + o) * Cin + kb + c;
            *(unsigned int*)&Bs[o][c] = pack_bf16x2(wr[0], wr[1]);
        }
        __syncthreads();

        // ---- fragments per ISA layout (16-bit A 16x32 / B 32x16, wave32) ----
        union Frag { u16x8 s[2]; bf16x16 v; };
        Frag fa[2], fb[4];
        const int khB = (lane >> 4) * 8;      // lanes 16-31: K {8..15, 24..31}
#pragma unroll
        for (int mt = 0; mt < 2; ++mt) {
            int m = mW + mt * 16 + (lane & 15);
            fa[mt].s[0] = *(const u16x8*)&As[m][khB];
            fa[mt].s[1] = *(const u16x8*)&As[m][khB + 16];
        }
        const int cB = (lane >> 4) * 8;       // lanes 16-31: K {8..15, 24..31}
#pragma unroll
        for (int nt = 0; nt < 4; ++nt) {
            int o = nW + nt * 16 + (lane & 15);
            fb[nt].s[0] = *(const u16x8*)&Bs[o][cB];
            fb[nt].s[1] = *(const u16x8*)&Bs[o][cB + 16];
        }
#pragma unroll
        for (int mt = 0; mt < 2; ++mt)
#pragma unroll
            for (int nt = 0; nt < 4; ++nt)
                acc[mt][nt] = __builtin_amdgcn_wmma_f32_16x16x32_bf16(
                    false, fa[mt].v, false, fb[nt].v, (short)0, acc[mt][nt],
                    false, false);
        __syncthreads();
    }

    // ---- epilogue: D element (row m, col n): vgpr j = m&7, lane = n + 16*(m>>3)
    const int mrowHi = 8 * (lane >> 4);
    const int ncol = lane & 15;
#pragma unroll
    for (int mt = 0; mt < 2; ++mt) {
#pragma unroll
        for (int nt = 0; nt < 4; ++nt) {
#pragma unroll
            for (int j = 0; j < 8; ++j) {
                int P = pTile + mW + mt * 16 + j + mrowHi;
                int O = oTile + nW + nt * 16 + ncol;
                float v = acc[mt][nt][j] + bias[O];
                size_t idx = ((size_t)b * Cout + O) * hw + P;
                if (resid) v += resid[idx];
                if (do_relu) v = fmaxf(v, 0.f);
                out[idx] = v;
            }
        }
    }
}

// ---------------------------------------------------------------------------
// Host orchestration
// ---------------------------------------------------------------------------
extern "C" void kernel_launch(void* const* d_in, const int* in_sizes, int n_in,
                              void* d_out, int out_size, void* d_ws, size_t ws_size,
                              hipStream_t stream) {
    (void)in_sizes; (void)n_in; (void)out_size; (void)ws_size;
    const int n = 4, C = 256, hw = 128 * 128;

    auto f = [&](int i) { return (const float*)d_in[i]; };
    const float* F = f(0);

    // workspace layout
    char* ws = (char*)d_ws;
    size_t off = 0;
    auto alloc = [&](size_t bytes) { void* p = ws + off; off += (bytes + 255) & ~(size_t)255; return p; };
    int*   rawMask = (int*)  alloc((size_t)n * hw * 4);
    float* fused1  = (float*)alloc((size_t)n * C * hw * 4);
    float* fused2  = (float*)alloc((size_t)n * C * hw * 4);
    float* fused3  = (float*)alloc((size_t)n * C * hw * 4);
    float* out1    = (float*)alloc((size_t)n * 512 * hw * 4);
    float* gsum    = (float*)alloc((size_t)n * 20 * 256 * 4);
    float* gmax    = (float*)alloc((size_t)n * 20 * 256 * 4);
    int*   gcnt    = (int*)  alloc((size_t)n * 20 * 4);
    float* X       = (float*)alloc((size_t)n * 20 * 512 * 4);
    float* h       = (float*)alloc((size_t)n * 512 * 20 * 4);
    float* hn      = (float*)alloc((size_t)n * 512 * 20 * 4);
    float* qb      = (float*)alloc((size_t)n * 128 * 20 * 4);
    float* kb      = (float*)alloc((size_t)n * 128 * 20 * 4);
    float* vb      = (float*)alloc((size_t)n * 512 * 20 * 4);
    float* gnode   = (float*)alloc((size_t)n * 256 * 20 * 4);

    float* OUT  = (float*)d_out;                         // (n,256,hw)
    float* xaux = OUT + (size_t)n * C * hw;              // (n,20,hw)

    // 1) semantic logits + argmax
    sem_argmax_kernel<<<(n * hw) / 256, 256, 0, stream>>>(F, f(1), f(2), xaux, rawMask,
                                                          hw, n * hw);

    const int lvBase[3] = {3, 34, 65};     // param-index base of lv1/lv2/lv3
    const int Ks[3]     = {2, 5, 20};
    const int divs[3]   = {10, 4, 1};      // LUT1: /10, LUT2: /4, raw: /1
    const int tW[3]     = {96, 98, 100};
    float* fusedArr[3]  = {fused1, fused2, fused3};

    const float* curIn = F;
    for (int L = 0; L < 3; ++L) {
        const int K = Ks[L], base = lvBase[L];
        const int totKC = n * K * 256, totK = n * K;

        agg_init_kernel<<<(totKC + 255) / 256, 256, 0, stream>>>(gsum, gmax, gcnt, totKC, totK);
        dim3 ag(hw / AGG_PIX, 1, n);
        agg_kernel<<<ag, 256, 0, stream>>>(curIn, rawMask, divs[L], K, gsum, gmax, gcnt, hw);

        int totX = n * K * 512;
        finalize_nodes_kernel<<<(totX + 255) / 256, 256, 0, stream>>>(gsum, gmax, gcnt, X, K, hw, totX);

        // h = gt_w @ X^T + gt_b        (n,512,K)
        int tot = n * 512 * K;
        node_gemm_kernel<<<(tot + 255) / 256, 256, 0, stream>>>(X, f(base + 0), f(base + 1),
                                                                h, 512, 512, K, 0, tot);
        for (int a = 0; a < 3; ++a) {
            bn_kernel<<<2, 256, 0, stream>>>(h, f(base + 25 + 2 * a), f(base + 26 + 2 * a), hn, K, n);
            int tq = n * 128 * K;
            node_gemm_kernel<<<(tq + 255) / 256, 256, 0, stream>>>(hn, f(base + 4 + 7 * a),
                f(base + 5 + 7 * a), qb, 512, 128, K, 1, tq);
            node_gemm_kernel<<<(tq + 255) / 256, 256, 0, stream>>>(hn, f(base + 6 + 7 * a),
                f(base + 7 + 7 * a), kb, 512, 128, K, 1, tq);
            int tv = n * 512 * K;
            node_gemm_kernel<<<(tv + 255) / 256, 256, 0, stream>>>(hn, f(base + 8 + 7 * a),
                f(base + 9 + 7 * a), vb, 512, 512, K, 1, tv);
            attn_kernel<<<n, 256, 0, stream>>>(qb, kb, vb, hn, f(base + 10 + 7 * a), h, K);
        }
        // g = gtb_w @ h + gtb_b        (n,256,K)
        int tg = n * 256 * K;
        node_gemm_kernel<<<(tg + 255) / 256, 256, 0, stream>>>(h, f(base + 2), f(base + 3),
                                                               gnode, 512, 256, K, 1, tg);

        // fusedL = curIn + conv_t(gather(g, mask))   — gather fused into WMMA A-load
        dim3 cg(hw / 64, 256 / 256, n);
        conv1x1_wmma_kernel<<<cg, 256, 0, stream>>>(gnode, nullptr, nullptr, nullptr, 256, 256,
            f(tW[L]), f(tW[L] + 1), 256, curIn, fusedArr[L], 0, rawMask, divs[L], K, hw);
        curIn = fusedArr[L];
    }

    // final: out1 = f1_w @ concat(F,fused1,fused2,fused3) + f1_b   (K=1024, 4 segments)
    dim3 cg1(hw / 64, 512 / 256, n);
    conv1x1_wmma_kernel<<<cg1, 256, 0, stream>>>(F, fused1, fused2, fused3, 1024, 256,
        f(102), f(103), 512, nullptr, out1, 0, nullptr, 1, 1, hw);
    // out = relu(f2_w @ out1 + f2_b)
    dim3 cg2(hw / 64, 256 / 256, n);
    conv1x1_wmma_kernel<<<cg2, 256, 0, stream>>>(out1, nullptr, nullptr, nullptr, 512, 512,
        f(104), f(105), 256, nullptr, OUT, 1, nullptr, 1, 1, hw);
}
